// CodeBook_12902081757285
// MI455X (gfx1250) — matrix operand order repaired and checked by reading, not compiled, not run on hardware
//
#include <hip/hip_runtime.h>

typedef __attribute__((ext_vector_type(16))) _Float16 v16h;
typedef __attribute__((ext_vector_type(8)))  float    v8f;

#define NCENTS 1024
#define DFILT  32
#define NROWS  (2048 * 2048 / DFILT)          // 131072 rows of 32
#define WAVES_PER_BLOCK 8
#define ROWS_PER_BLOCK  (WAVES_PER_BLOCK * 16) // 128
#define NBLOCKS         (NROWS / ROWS_PER_BLOCK) // 1024
#define NTILES          (NCENTS / 16)            // 64
#define CSTRIDE_H 40                 // centroid row stride in halfs (32 + 8 pad)
#define CSTRIDE_B (CSTRIDE_H * 2)    // 80 bytes -> conflict-free b128 B-frag loads

// ---------------------------------------------------------------------------
// Kernel 1: new_lat = lat * clip(leak, 0.001, 1000)   (pure bandwidth)
// ---------------------------------------------------------------------------
__global__ void vq_scale(const float4* __restrict__ lat,
                         const float*  __restrict__ leak,
                         float4*       __restrict__ out) {
  int i = blockIdx.x * blockDim.x + threadIdx.x;
  float lf = fminf(fmaxf(leak[0], 0.001f), 1000.0f);
  float4 v = lat[i];
  v.x *= lf; v.y *= lf; v.z *= lf; v.w *= lf;
  out[i] = v;
}

// ---------------------------------------------------------------------------
// Kernel 2: per-wave 16-row tile scans all 1024 centroids with f16 WMMA
// (hi/lo split: score = ahi*bhi + ahi*blo + alo*bhi ~ f32 accuracy).
// Two independent accumulator chains per iteration so the 3-WMMA chain of
// one tile and the 8 min-update VALU ops of the other fill each other's
// WMMA->VALU hazard slots (no v_nop stalls).
// ---------------------------------------------------------------------------
__global__ __launch_bounds__(256) void vq_min(const float* __restrict__ lat,
                                              const float* __restrict__ cent,
                                              float* __restrict__ partial) {
  extern __shared__ char smem[];
  _Float16* bhi  = (_Float16*)(smem);                          // 1024*40 halfs
  _Float16* blo  = (_Float16*)(smem + NCENTS * CSTRIDE_B);     // 1024*40 halfs
  float*    cnsh = (float*)   (smem + 2 * NCENTS * CSTRIDE_B); // 1024 f32
  float*    znsh = (float*)   (smem + 2 * NCENTS * CSTRIDE_B + NCENTS * 4); // 8*16

  const int tid = threadIdx.x;

  // ---- stage centroids into LDS as f16 hi/lo + norms (padded rows) ----
  for (int c = tid; c < NCENTS; c += 256) {
    const float* cp = cent + c * DFILT;
    float nrm = 0.0f;
#pragma unroll
    for (int k = 0; k < DFILT; ++k) {
      float x = cp[k];
      nrm += x * x;
      _Float16 h = (_Float16)x;
      bhi[c * CSTRIDE_H + k] = h;
      blo[c * CSTRIDE_H + k] = (_Float16)(x - (float)h);
    }
    cnsh[c] = nrm;
  }
  __syncthreads();

  const int wv     = tid >> 5;
  const int lane   = tid & 31;
  const int hiHalf = lane >> 4;         // 0: lanes 0-15, 1: lanes 16-31
  const int l16    = lane & 15;
  const int row0   = (blockIdx.x * WAVES_PER_BLOCK + wv) * 16;
  const float* zrow = lat + (size_t)(row0 + l16) * DFILT;

  // ---- build A fragment (16x32, contiguous k order k = 16*half + s, as B) ----
  // lanes<16: K = 0..15 ; lanes>=16: K = 16..31
  const int o = hiHalf * 16;
  float f[16];
#pragma unroll
  for (int j = 0; j < 8; ++j) f[j]     = zrow[o + j];
#pragma unroll
  for (int j = 0; j < 8; ++j) f[8 + j] = zrow[o + 8 + j];

  v16h ahi, alo;
  float zn = 0.0f;
#pragma unroll
  for (int j = 0; j < 16; ++j) {
    float x = f[j];
    zn += x * x;
    _Float16 h = (_Float16)x;
    ahi[j] = h;
    alo[j] = (_Float16)(x - (float)h);
  }
  zn += __shfl_xor(zn, 16, 32);          // lane L and L^16 hold same row
  if (lane < 16) znsh[wv * 16 + lane] = zn;

  float mn[8];
#pragma unroll
  for (int r = 0; r < 8; ++r) mn[r] = 3.4e38f;

  const int koff = hiHalf * 32;          // byte offset: 16 halfs into centroid row

  // ---- sweep 64 centroid tiles, two at a time (dual accumulator chains) ----
  for (int t = 0; t < NTILES; t += 2) {
    const int n0 = t * 16 + l16;
    const int n1 = n0 + 16;
    const v16h bh0 = *(const v16h*)((const char*)bhi + n0 * CSTRIDE_B + koff);
    const v16h bl0 = *(const v16h*)((const char*)blo + n0 * CSTRIDE_B + koff);
    const v16h bh1 = *(const v16h*)((const char*)bhi + n1 * CSTRIDE_B + koff);
    const v16h bl1 = *(const v16h*)((const char*)blo + n1 * CSTRIDE_B + koff);
    const float cn0 = cnsh[n0];
    const float cn1 = cnsh[n1];

    v8f acc0 = {};
    acc0 = __builtin_amdgcn_wmma_f32_16x16x32_f16(false, alo, false, bh0,
                                                  (short)0, acc0, false, false);
    acc0 = __builtin_amdgcn_wmma_f32_16x16x32_f16(false, ahi, false, bl0,
                                                  (short)0, acc0, false, false);
    acc0 = __builtin_amdgcn_wmma_f32_16x16x32_f16(false, ahi, false, bh0,
                                                  (short)0, acc0, false, false);
    v8f acc1 = {};
    acc1 = __builtin_amdgcn_wmma_f32_16x16x32_f16(false, alo, false, bh1,
                                                  (short)0, acc1, false, false);
    acc1 = __builtin_amdgcn_wmma_f32_16x16x32_f16(false, ahi, false, bl1,
                                                  (short)0, acc1, false, false);
    // consume acc0 between the two chains: fills acc0's WMMA->VALU hazard
    float d0[8];
#pragma unroll
    for (int r = 0; r < 8; ++r) d0[r] = fmaf(-2.0f, acc0[r], cn0);
    acc1 = __builtin_amdgcn_wmma_f32_16x16x32_f16(false, ahi, false, bh1,
                                                  (short)0, acc1, false, false);
#pragma unroll
    for (int r = 0; r < 8; ++r) mn[r] = fminf(mn[r], d0[r]);
#pragma unroll
    for (int r = 0; r < 8; ++r) {
      float d1 = fmaf(-2.0f, acc1[r], cn1);
      mn[r] = fminf(mn[r], d1);
    }
  }

  // ---- min across the 16 columns (lanes within each half) ----
#pragma unroll
  for (int r = 0; r < 8; ++r) {
#pragma unroll
    for (int off = 1; off < 16; off <<= 1)
      mn[r] = fminf(mn[r], __shfl_xor(mn[r], off, 32));
  }

  // slot r in low half = row r, in high half = row r+8
  float s = 0.0f;
#pragma unroll
  for (int r = 0; r < 8; ++r)
    s += mn[r] + znsh[wv * 16 + hiHalf * 8 + r];
  s += __shfl_xor(s, 16, 32);            // combine both row halves
  if (lane == 0)
    partial[blockIdx.x * WAVES_PER_BLOCK + wv] = s;
}

// ---------------------------------------------------------------------------
// Kernel 3: deterministic fixed-order reduction of 8192 wave partials
// ---------------------------------------------------------------------------
__global__ void vq_reduce(const float* __restrict__ partial,
                          float* __restrict__ loss_out) {
  __shared__ float sm[1024];
  int tid = threadIdx.x;
  float s = 0.0f;
  for (int i = tid; i < NBLOCKS * WAVES_PER_BLOCK; i += 1024) s += partial[i];
  sm[tid] = s;
  __syncthreads();
  for (int st = 512; st > 0; st >>= 1) {
    if (tid < st) sm[tid] += sm[tid + st];
    __syncthreads();
  }
  if (tid == 0)
    loss_out[0] = sm[0] * (1.25f / (float)(NROWS * DFILT)); // (1+BETA)*mean
}

extern "C" void kernel_launch(void* const* d_in, const int* in_sizes, int n_in,
                              void* d_out, int out_size, void* d_ws, size_t ws_size,
                              hipStream_t stream) {
  const float* lat  = (const float*)d_in[0];   // [2048*2048]
  const float* cent = (const float*)d_in[1];   // [1024*32]
  const float* leak = (const float*)d_in[2];   // [1]
  float* out     = (float*)d_out;              // [2048*2048 + 1]
  float* partial = (float*)d_ws;               // 8192 floats scratch

  // output 0: scaled lat
  vq_scale<<<(2048 * 2048 / 4) / 256, 256, 0, stream>>>(
      (const float4*)lat, leak, (float4*)out);

  // output 1: VQ loss
  size_t smemBytes = 2 * NCENTS * CSTRIDE_B + NCENTS * 4 +
                     WAVES_PER_BLOCK * 16 * 4;               // 168,448 B LDS
  vq_min<<<NBLOCKS, 256, smemBytes, stream>>>(lat, cent, partial);
  vq_reduce<<<1, 1024, 0, stream>>>(partial, out + 2048 * 2048);
}
